// SpatialGATLarge_90795608637705
// MI455X (gfx1250) — matrix operand order repaired and checked by reading, not compiled, not run on hardware
//
#include <hip/hip_runtime.h>
#include <hip/hip_bf16.h>
#include <cmath>
#include <cstdint>

// ---------------------------------------------------------------------------
// Types for CDNA5 WMMA (wave32)
// ---------------------------------------------------------------------------
typedef __attribute__((ext_vector_type(16))) __bf16          v16bf;
typedef __attribute__((ext_vector_type(16))) unsigned short  v16us;
typedef __attribute__((ext_vector_type(8)))  float           v8f;

#define NEG_SLOPE 0.2f
#define GAT_EPS   1e-16f

// ---------------------------------------------------------------------------
// Helpers
// ---------------------------------------------------------------------------
static __device__ __forceinline__ unsigned short f32_to_bf16_rne(float f) {
  unsigned int u = __float_as_uint(f);
  u += 0x7fffu + ((u >> 16) & 1u);   // round-to-nearest-even
  return (unsigned short)(u >> 16);
}

static __device__ __forceinline__ unsigned int pack2_bf16(float lo, float hi) {
  return (unsigned int)f32_to_bf16_rne(lo) | ((unsigned int)f32_to_bf16_rne(hi) << 16);
}

static __device__ __forceinline__ void edge_sd(const int* __restrict__ ei,
                                               int E, int e, int& s, int& d) {
  if (e < E) { s = ei[e]; d = ei[E + e]; }
  else       { s = e - E; d = e - E; }         // self-loop
}

static __device__ __forceinline__ void atomic_max_f32(float* addr, float v) {
  if (v >= 0.0f) atomicMax((int*)addr, __float_as_int(v));
  else           atomicMin((unsigned int*)addr, __float_as_uint(v));
}

// ---------------------------------------------------------------------------
// Elementwise kernels
// ---------------------------------------------------------------------------
__global__ void fill_kernel(float* __restrict__ p, float v, int n) {
  int i = blockIdx.x * blockDim.x + threadIdx.x;
  if (i < n) p[i] = v;
}

// out += bias (broadcast over rows), then ELU, in place
__global__ void bias_elu_kernel(float* __restrict__ io, const float* __restrict__ b,
                                int Nn, int F) {
  int i = blockIdx.x * blockDim.x + threadIdx.x;
  if (i >= Nn * F) return;
  float v = io[i] + b[i % F];
  io[i] = v > 0.0f ? v : (__expf(v) - 1.0f);
}

// ---------------------------------------------------------------------------
// Fragment packing: fp32 -> bf16 while permuting into per-lane WMMA order.
//
// A fragment (16x32 bf16), lane = half*16 + row:
//   VGPR j (j<4): K = k0 + 8*half + 2j ; (j>=4): K = k0 + 8*half + 16 + 2(j-4)
// Apacked layout: [mtile][kt][lane][8 dwords]  (one dword = 2 packed bf16)
// ---------------------------------------------------------------------------
__global__ void packA_kernel(const float* __restrict__ A,  // [M,K] f32
                             unsigned int* __restrict__ P, // M16*Kt*256 dwords
                             int M, int K) {
  int i = blockIdx.x * blockDim.x + threadIdx.x;
  const int Kt  = K >> 5;
  const int M16 = (M + 15) >> 4;
  if (i >= M16 * Kt * 256) return;
  int within = i & 255;           // dword within (mtile,kt) block
  int blk    = i >> 8;
  int j2     = within & 7;        // VGPR index 0..7
  int lane   = within >> 3;       // 0..31
  int kt     = blk % Kt;
  int tile   = blk / Kt;
  int half   = lane >> 4;
  int row    = tile * 16 + (lane & 15);
  int k      = kt * 32 + 8 * half + ((j2 < 4) ? (2 * j2) : (16 + 2 * (j2 - 4)));
  float f0 = 0.f, f1 = 0.f;
  if (row < M) {
    const float* ap = A + (size_t)row * K + k;
    f0 = ap[0]; f1 = ap[1];
  }
  P[i] = pack2_bf16(f0, f1);
}

// ---------------------------------------------------------------------------
// B fragment (32x16 bf16), lane = half*16 + col:
//   VGPR j (j<4): K = k0 + 8*half + 2j ; (j>=4): K = k0 + 8*half + 16 + 2(j-4)
// Bpacked layout: [ntile][kt][lane][8 dwords]
// ---------------------------------------------------------------------------
__global__ void packB_kernel(const float* __restrict__ B,  // [K,N] f32
                             unsigned int* __restrict__ P, // (N/16)*Kt*256 dwords
                             int K, int N) {
  int i = blockIdx.x * blockDim.x + threadIdx.x;
  const int Kt  = K >> 5;
  const int N16 = N >> 4;
  if (i >= N16 * Kt * 256) return;
  int within = i & 255;
  int blk    = i >> 8;
  int j2     = within & 7;
  int lane   = within >> 3;
  int kt     = blk % Kt;
  int nt     = blk / Kt;
  int half   = lane >> 4;
  int col    = nt * 16 + (lane & 15);
  int k      = kt * 32 + 8 * half + ((j2 < 4) ? (2 * j2) : (16 + 2 * (j2 - 4)));
  P[i] = pack2_bf16(B[(size_t)k * N + col], B[(size_t)(k + 1) * N + col]);
}

// ---------------------------------------------------------------------------
// WMMA bf16 GEMM on pre-packed fragments: C[M,N] = A[M,K] * B[K,N]
// Block = 4 waves; each wave computes a 32x64 tile (2 M-tiles x 4 N-tiles,
// B fragments reused across both M-tiles -> 8 WMMAs per k-step).
// Double-buffered by parity (kt & 1) + unroll-by-2, so the stage index is a
// compile-time constant in each unrolled body: no register rotation copies.
// ---------------------------------------------------------------------------
__global__ __launch_bounds__(128)
void wmma_gemm_packed_kernel(const unsigned int* __restrict__ Ap,
                             const unsigned int* __restrict__ Bp,
                             float* __restrict__ C,
                             int M, int N, int K) {
  const int Kt    = K >> 5;
  const int M16   = (M + 15) >> 4;
  const int lane  = threadIdx.x & 31;
  const int wave  = threadIdx.x >> 5;
  const int mt0   = blockIdx.x * 8 + wave * 2;   // first of 2 M-tiles
  if (mt0 >= M16) return;
  const bool has1 = (mt0 + 1) < M16;
  const int n0t   = blockIdx.y * 4;              // first of 4 N-tiles

  const unsigned int* a0 = Ap + ((size_t)mt0 * Kt) * 256 + lane * 8;
  const unsigned int* a1 = has1 ? (Ap + ((size_t)(mt0 + 1) * Kt) * 256 + lane * 8) : a0;
  const unsigned int* bb[4];
#pragma unroll
  for (int t = 0; t < 4; ++t)
    bb[t] = Bp + ((size_t)(n0t + t) * Kt) * 256 + lane * 8;

  v8f acc[8];
#pragma unroll
  for (int t = 0; t < 8; ++t) acc[t] = v8f{0.f,0.f,0.f,0.f,0.f,0.f,0.f,0.f};

  // stage buffers: [parity][...]
  v16us Ab[2][2];
  v16us Bb[2][4];
  Ab[0][0] = *(const v16us*)a0;
  Ab[0][1] = *(const v16us*)a1;
#pragma unroll
  for (int t = 0; t < 4; ++t) Bb[0][t] = *(const v16us*)bb[t];

#pragma unroll 2
  for (int kt = 0; kt < Kt; ++kt) {
    const int cur = kt & 1;
    const int nxt = cur ^ 1;
    if (kt + 1 < Kt) {                        // prefetch next k-step into nxt
      size_t o = (size_t)(kt + 1) * 256;
      Ab[nxt][0] = *(const v16us*)(a0 + o);
      Ab[nxt][1] = *(const v16us*)(a1 + o);
#pragma unroll
      for (int t = 0; t < 4; ++t) Bb[nxt][t] = *(const v16us*)(bb[t] + o);
    }
    v16bf fa0 = __builtin_bit_cast(v16bf, Ab[cur][0]);
    v16bf fa1 = __builtin_bit_cast(v16bf, Ab[cur][1]);
#pragma unroll
    for (int t = 0; t < 4; ++t) {
      v16bf fb = __builtin_bit_cast(v16bf, Bb[cur][t]);
      acc[t]     = __builtin_amdgcn_wmma_f32_16x16x32_bf16(
          false, fa0, false, fb, (short)0, acc[t],     false, false);
      acc[4 + t] = __builtin_amdgcn_wmma_f32_16x16x32_bf16(
          false, fa1, false, fb, (short)0, acc[4 + t], false, false);
    }
  }

  // C/D layout: lane = half*16 + col ; VGPR j -> row = 8*half + j
  const int half = lane >> 4;
  const int lr   = lane & 15;
#pragma unroll
  for (int t = 0; t < 4; ++t) {
    int col = (n0t + t) * 16 + lr;
#pragma unroll
    for (int j = 0; j < 8; ++j) {
      int row0 = mt0 * 16 + 8 * half + j;
      if (row0 < M) C[(size_t)row0 * N + col] = acc[t][j];
      if (has1) {
        int row1 = (mt0 + 1) * 16 + 8 * half + j;
        if (row1 < M) C[(size_t)row1 * N + col] = acc[4 + t][j];
      }
    }
  }
}

// ---------------------------------------------------------------------------
// Attention scalar products: aS[n,h] = sum_c h[n,h,c]*att_s[h,c] (same for aD)
// ---------------------------------------------------------------------------
__global__ void attn_dot_kernel(const float* __restrict__ h,
                                const float* __restrict__ att_s,
                                const float* __restrict__ att_d,
                                float* __restrict__ aS, float* __restrict__ aD,
                                int Nn, int H, int C) {
  int i = blockIdx.x * blockDim.x + threadIdx.x;
  if (i >= Nn * H) return;
  int n = i / H, hh = i % H;
  const float* hp = h + (size_t)n * H * C + (size_t)hh * C;
  const float* as = att_s + (size_t)hh * C;
  const float* ad = att_d + (size_t)hh * C;
  float ss = 0.f, dd = 0.f;
  for (int c = 0; c < C; c += 4) {
    float4 hv = *(const float4*)(hp + c);
    float4 sv = *(const float4*)(as + c);
    float4 dv = *(const float4*)(ad + c);
    ss += hv.x * sv.x + hv.y * sv.y + hv.z * sv.z + hv.w * sv.w;
    dd += hv.x * dv.x + hv.y * dv.y + hv.z * dv.z + hv.w * dv.w;
  }
  aS[i] = ss;
  aD[i] = dd;
}

// ---------------------------------------------------------------------------
// Edge softmax pass 1: per-destination running max of leaky_relu(aS[s]+aD[d])
// ---------------------------------------------------------------------------
__global__ void edge_max_kernel(const int* __restrict__ ei, int E, int Etot,
                                const float* __restrict__ aS, const float* __restrict__ aD,
                                float* __restrict__ mB, int H) {
  int i = blockIdx.x * blockDim.x + threadIdx.x;
  if (i >= Etot * H) return;
  int e = i / H, hh = i % H;
  int s, d; edge_sd(ei, E, e, s, d);
  float v = aS[(size_t)s * H + hh] + aD[(size_t)d * H + hh];
  v = v > 0.f ? v : NEG_SLOPE * v;
  atomic_max_f32(&mB[(size_t)d * H + hh], v);
}

// Pass 2: e = exp(v - m[dst]) ; eexp[e,h] = e ; s[dst,h] += e
__global__ void edge_exp_kernel(const int* __restrict__ ei, int E, int Etot,
                                const float* __restrict__ aS, const float* __restrict__ aD,
                                const float* __restrict__ mB,
                                float* __restrict__ eexp, float* __restrict__ sB, int H) {
  int i = blockIdx.x * blockDim.x + threadIdx.x;
  if (i >= Etot * H) return;
  int e = i / H, hh = i % H;
  int s, d; edge_sd(ei, E, e, s, d);
  float v = aS[(size_t)s * H + hh] + aD[(size_t)d * H + hh];
  v = v > 0.f ? v : NEG_SLOPE * v;
  float ex = __expf(v - mB[(size_t)d * H + hh]);
  eexp[(size_t)e * H + hh] = ex;
  atomicAdd(&sB[(size_t)d * H + hh], ex);
}

// Pass 3: out[dst,h,c] += h[src,h,c] * alpha ; one thread per (edge, head, chan4)
__global__ void edge_msg_kernel(const int* __restrict__ ei, int E, int Etot,
                                const float* __restrict__ hfeat,
                                const float* __restrict__ eexp, const float* __restrict__ sB,
                                float* __restrict__ out, int H, int C) {
  long long i = (long long)blockIdx.x * blockDim.x + threadIdx.x;
  int Cq = C >> 2;
  long long total = (long long)Etot * H * Cq;
  if (i >= total) return;
  int q  = (int)(i % Cq);
  long long t = i / Cq;
  int hh = (int)(t % H);
  int e  = (int)(t / H);
  int s, d; edge_sd(ei, E, e, s, d);
  float alpha = eexp[(size_t)e * H + hh] / (sB[(size_t)d * H + hh] + GAT_EPS);
  const float4 hv = *(const float4*)(hfeat + (size_t)s * H * C + (size_t)hh * C + 4 * q);
  float* op = out + (size_t)d * H * C + (size_t)hh * C + 4 * q;
  atomicAdd(op + 0, hv.x * alpha);
  atomicAdd(op + 1, hv.y * alpha);
  atomicAdd(op + 2, hv.z * alpha);
  atomicAdd(op + 3, hv.w * alpha);
}

// ---------------------------------------------------------------------------
// MLP: hidden = relu(in[N,64] @ W[64,32] + b) ; out = hidden @ W2[32,2] + b2
// ---------------------------------------------------------------------------
__global__ void mlp1_kernel(const float* __restrict__ in, const float* __restrict__ W,
                            const float* __restrict__ b, float* __restrict__ out, int Nn) {
  int i = blockIdx.x * blockDim.x + threadIdx.x;
  if (i >= Nn * 32) return;
  int n = i / 32, j = i % 32;
  const float* ip = in + (size_t)n * 64;
  float acc = b[j];
  for (int k = 0; k < 64; ++k) acc += ip[k] * W[k * 32 + j];
  out[i] = fmaxf(acc, 0.f);
}

__global__ void mlp2_kernel(const float* __restrict__ hdn, const float* __restrict__ W,
                            const float* __restrict__ b, float* __restrict__ out, int Nn) {
  int n = blockIdx.x * blockDim.x + threadIdx.x;
  if (n >= Nn) return;
  const float* hp = hdn + (size_t)n * 32;
  float a0 = b[0], a1 = b[1];
  for (int k = 0; k < 32; ++k) {
    float v = hp[k];
    a0 += v * W[k * 2 + 0];
    a1 += v * W[k * 2 + 1];
  }
  out[(size_t)n * 2 + 0] = a0;
  out[(size_t)n * 2 + 1] = a1;
}

// ---------------------------------------------------------------------------
// Host-side: one GAT layer
// ---------------------------------------------------------------------------
static void run_gat_layer(hipStream_t stream,
                          const float* inFeat, int Nn, int Fi, int H, int C,
                          const float* W, const float* bias,
                          const float* attS, const float* attD,
                          const int* ei, int E, int Etot,
                          unsigned int* Apacked, unsigned int* Bpacked,
                          float* bufH, float* bufO,
                          float* aS, float* aD, float* mB, float* sB, float* eexp) {
  const int Fo = H * C;
  const int TB = 256;
  const int Kt  = Fi >> 5;
  const int M16 = (Nn + 15) >> 4;

  // pack A (activations) and B (weights) into WMMA fragment order (+ bf16 cast)
  int nA = M16 * Kt * 256;
  packA_kernel<<<(nA + TB - 1) / TB, TB, 0, stream>>>(inFeat, Apacked, Nn, Fi);
  int nB = (Fo >> 4) * Kt * 256;
  packB_kernel<<<(nB + TB - 1) / TB, TB, 0, stream>>>(W, Bpacked, Fi, Fo);

  // each block: 4 waves x (2 M-tiles each) = 8 M-tiles; grid.y covers N in 64s
  dim3 gg((M16 + 7) / 8, Fo / 64);
  wmma_gemm_packed_kernel<<<gg, 128, 0, stream>>>(Apacked, Bpacked, bufH, Nn, Fo, Fi);

  int nNH = Nn * H;
  attn_dot_kernel<<<(nNH + TB - 1) / TB, TB, 0, stream>>>(bufH, attS, attD, aS, aD, Nn, H, C);

  fill_kernel<<<(nNH + TB - 1) / TB, TB, 0, stream>>>(mB, -INFINITY, nNH);
  fill_kernel<<<(nNH + TB - 1) / TB, TB, 0, stream>>>(sB, 0.f, nNH);
  int nO = Nn * Fo;
  fill_kernel<<<(nO + TB - 1) / TB, TB, 0, stream>>>(bufO, 0.f, nO);

  int nEH = Etot * H;
  edge_max_kernel<<<(nEH + TB - 1) / TB, TB, 0, stream>>>(ei, E, Etot, aS, aD, mB, H);
  edge_exp_kernel<<<(nEH + TB - 1) / TB, TB, 0, stream>>>(ei, E, Etot, aS, aD, mB, eexp, sB, H);

  long long tot = (long long)Etot * H * (C / 4);
  unsigned int gmsg = (unsigned int)((tot + TB - 1) / TB);
  edge_msg_kernel<<<gmsg, TB, 0, stream>>>(ei, E, Etot, bufH, eexp, sB, bufO, H, C);

  bias_elu_kernel<<<(nO + TB - 1) / TB, TB, 0, stream>>>(bufO, bias, Nn, Fo);
}

// ---------------------------------------------------------------------------
// Entry point
// ---------------------------------------------------------------------------
extern "C" void kernel_launch(void* const* d_in, const int* in_sizes, int n_in,
                              void* d_out, int out_size, void* d_ws, size_t ws_size,
                              hipStream_t stream) {
  const float* x   = (const float*)d_in[0];
  const int*   ei  = (const int*)d_in[1];
  const float* W1  = (const float*)d_in[2];
  const float* b1  = (const float*)d_in[3];
  const float* as1 = (const float*)d_in[4];
  const float* ad1 = (const float*)d_in[5];
  const float* W2  = (const float*)d_in[6];
  const float* b2  = (const float*)d_in[7];
  const float* as2 = (const float*)d_in[8];
  const float* ad2 = (const float*)d_in[9];
  const float* W3  = (const float*)d_in[10];
  const float* b3  = (const float*)d_in[11];
  const float* as3 = (const float*)d_in[12];
  const float* ad3 = (const float*)d_in[13];
  const float* Wm1 = (const float*)d_in[14];
  const float* bm1 = (const float*)d_in[15];
  const float* Wm2 = (const float*)d_in[16];
  const float* bm2 = (const float*)d_in[17];

  const int Nn   = in_sizes[0] / 64;   // 50000
  const int E    = in_sizes[1] / 2;    // 400000
  const int Etot = E + Nn;             // + self loops
  const int FMAX = 512;                // widest feature dim
  const int HMAX = 4;

  // ---- workspace layout
  char*  ws  = (char*)d_ws;
  size_t off = 0;
  auto alloc = [&](size_t bytes) -> void* {
    void* p = ws + off;
    off = (off + bytes + 255) & ~(size_t)255;
    return p;
  };
  unsigned int* Apacked = (unsigned int*)alloc((size_t)Nn * FMAX * 2);      // bf16 = half the f32 bytes
  unsigned int* Bpacked = (unsigned int*)alloc((size_t)512 * 192 * 2);
  float* bufH = (float*)alloc((size_t)Nn * FMAX * 4);
  float* bufO = (float*)alloc((size_t)Nn * FMAX * 4);
  float* aS   = (float*)alloc((size_t)Nn * HMAX * 4);
  float* aD   = (float*)alloc((size_t)Nn * HMAX * 4);
  float* mB   = (float*)alloc((size_t)Nn * HMAX * 4);
  float* sB   = (float*)alloc((size_t)Nn * HMAX * 4);
  float* eexp = (float*)alloc((size_t)Etot * HMAX * 4);

  // ---- GAT layers (bufO ping-pongs as next layer's f32 input)
  run_gat_layer(stream, x,    Nn,  64, 4, 128, W1, b1, as1, ad1, ei, E, Etot,
                Apacked, Bpacked, bufH, bufO, aS, aD, mB, sB, eexp);
  run_gat_layer(stream, bufO, Nn, 512, 2,  96, W2, b2, as2, ad2, ei, E, Etot,
                Apacked, Bpacked, bufH, bufO, aS, aD, mB, sB, eexp);
  run_gat_layer(stream, bufO, Nn, 192, 1,  64, W3, b3, as3, ad3, ei, E, Etot,
                Apacked, Bpacked, bufH, bufO, aS, aD, mB, sB, eexp);

  // ---- MLP head: bufO[N,64] -> bufH[N,32] -> d_out[N,2]
  const int TB = 256;
  int n1 = Nn * 32;
  mlp1_kernel<<<(n1 + TB - 1) / TB, TB, 0, stream>>>(bufO, Wm1, bm1, bufH, Nn);
  mlp2_kernel<<<(Nn + TB - 1) / TB, TB, 0, stream>>>(bufH, Wm2, bm2, (float*)d_out, Nn);
}
